// FeatureModel_22436909154420
// MI455X (gfx1250) — matrix operand order repaired and checked by reading, not compiled, not run on hardware
//
#include <hip/hip_runtime.h>
#include <hip/hip_bf16.h>
#include <math.h>

// ---------------- model constants ----------------
#define N_SPECIES 119
#define N_RADIAL  5
#define N_BASIS   7
#define N_ATOMS   20000
#define N_PAIRS   800000
#define N_FEAT    360
#define K1_PAD    384          // 360 padded to multiple of 32 for WMMA K
#define UNITS     512

#define EMB_NORM  0.3779644730f     // 1/sqrt(7)
#define BETTA     (49.0f/36.0f)     // N_BASIS^2 / R_MAX^2
#define RAD_NORM  0.8981178f        // (2*betta/pi)^0.75
#define RSQ360    0.0527046277f     // 1/sqrt(360)
#define RSQ512    0.0441941738f     // 1/sqrt(512)
#define PI_F      3.14159265358979f

// per-atom symmetric moment storage: per radial r (5): [M0(1), M1(3), M2sym(6), M3sym(10)] = 20
#define MOM_PER_ATOM 100

typedef __attribute__((ext_vector_type(16))) _Float16 v16h;
typedef __attribute__((ext_vector_type(8)))  float    v8f;
typedef unsigned int u32x4 __attribute__((ext_vector_type(4)));
typedef int          i32x4 __attribute__((ext_vector_type(4)));
typedef int          i32x8 __attribute__((ext_vector_type(8)));

union Frag16 { v16h v; uint4 q[2]; };

// symmetric index tables
__device__ __constant__ int   kS2[9]  = {0,1,2, 1,3,4, 2,4,5};
__device__ __constant__ int   kS3[27] = {0,1,2,1,3,4,2,4,5,
                                         1,3,4,3,6,7,4,7,8,
                                         2,4,5,4,7,8,5,8,9};
__device__ __constant__ float kW2[6]  = {1,2,2,1,2,1};
__device__ __constant__ float kW3[10] = {1,3,3,3,6,3,1,3,3,1};

__device__ inline float m2g(const float M2s[N_RADIAL][6], int r, int i, int j) {
    return M2s[r][kS2[i*3 + j]];
}
__device__ inline float m3g(const float M3s[N_RADIAL][10], int r, int i, int j, int k) {
    return M3s[r][kS3[(i*3 + j)*3 + k]];
}

// ---------------- kernel 0: zero the moment accumulator ----------------
__global__ void mtp_zero_kernel(float* __restrict__ p, int n) {
    int i = blockIdx.x * blockDim.x + threadIdx.x;
    if (i < n) p[i] = 0.0f;
}

// ---------------- kernel 1: pair stage (gather + symmetric moment atomics) ----------------
__global__ __launch_bounds__(256) void mtp_pair_kernel(
    const float* __restrict__ R, const int* __restrict__ Z,
    const int* __restrict__ nbr, const float* __restrict__ offs,
    const float* __restrict__ emb, float* __restrict__ moments)
{
    int p = blockIdx.x * blockDim.x + threadIdx.x;
    if (p >= N_PAIRS) return;

    int i = nbr[p];
    int j = nbr[N_PAIRS + p];

    float dx = R[3*j+0] - R[3*i+0] + offs[3*p+0];
    float dy = R[3*j+1] - R[3*i+1] + offs[3*p+1];
    float dz = R[3*j+2] - R[3*i+2] + offs[3*p+2];
    float dr = sqrtf(dx*dx + dy*dy + dz*dz);
    float inv = 1.0f / (dr + 1e-5f);
    float d[3] = {dx*inv, dy*inv, dz*inv};

    float basis[N_BASIS];
    #pragma unroll
    for (int b = 0; b < N_BASIS; ++b) {
        float shift = 0.5f + (float)b * (5.5f / 6.0f);
        float t = dr - shift;
        basis[b] = RAD_NORM * __expf(-BETTA * t * t);
    }
    float cutoff = (dr < 6.0f) ? 0.5f * (__cosf(PI_F * dr / 6.0f) + 1.0f) : 0.0f;

    int zi = Z[i], zj = Z[j];
    const float* e = emb + (size_t)(zj * N_SPECIES + zi) * (N_RADIAL * N_BASIS);

    // unique direction products (symmetric storage, sorted-index order)
    float p2[6]  = {d[0]*d[0], d[0]*d[1], d[0]*d[2], d[1]*d[1], d[1]*d[2], d[2]*d[2]};
    float p3[10] = {p2[0]*d[0], p2[0]*d[1], p2[0]*d[2], p2[3]*d[0], p2[1]*d[2],
                    p2[5]*d[0], p2[3]*d[1], p2[3]*d[2], p2[5]*d[1], p2[5]*d[2]};

    float* mom = moments + (size_t)i * MOM_PER_ATOM;
    #pragma unroll
    for (int r = 0; r < N_RADIAL; ++r) {
        float rad = 0.0f;
        #pragma unroll
        for (int b = 0; b < N_BASIS; ++b) rad += e[r*N_BASIS + b] * basis[b];
        rad *= EMB_NORM * cutoff;
        float* m = mom + r * 20;
        atomicAdd(m + 0, rad);
        #pragma unroll
        for (int t = 0; t < 3;  ++t) atomicAdd(m + 1  + t, rad * d[t]);
        #pragma unroll
        for (int t = 0; t < 6;  ++t) atomicAdd(m + 4  + t, rad * p2[t]);
        #pragma unroll
        for (int t = 0; t < 10; ++t) atomicAdd(m + 10 + t, rad * p3[t]);
    }
}

// ---------------- kernel 2: per-atom tensor contractions -> 360 feats (f16, padded to 384) ----------------
__global__ __launch_bounds__(128) void mtp_contract_kernel(
    const float* __restrict__ moments, _Float16* __restrict__ feats)
{
    int a = blockIdx.x * blockDim.x + threadIdx.x;
    if (a >= N_ATOMS) return;

    const float* mom = moments + (size_t)a * MOM_PER_ATOM;
    float M0v[N_RADIAL], M1v[N_RADIAL][3], M2s[N_RADIAL][6], M3s[N_RADIAL][10];
    #pragma unroll
    for (int r = 0; r < N_RADIAL; ++r) {
        const float* m = mom + r * 20;
        M0v[r] = m[0];
        #pragma unroll
        for (int t = 0; t < 3;  ++t) M1v[r][t] = m[1 + t];
        #pragma unroll
        for (int t = 0; t < 6;  ++t) M2s[r][t] = m[4 + t];
        #pragma unroll
        for (int t = 0; t < 10; ++t) M3s[r][t] = m[10 + t];
    }

    _Float16* F = feats + (size_t)a * K1_PAD;
    int f = 0;

    // M0 (5)
    for (int r = 0; r < N_RADIAL; ++r) F[f++] = (_Float16)M0v[r];

    // c1: M1.M1 tril (15)
    #pragma unroll 1
    for (int r = 0; r < N_RADIAL; ++r)
    for (int s = 0; s <= r; ++s) {
        float acc = 0.0f;
        for (int dd = 0; dd < 3; ++dd) acc += M1v[r][dd] * M1v[s][dd];
        F[f++] = (_Float16)acc;
    }
    // c2: M2:M2 tril (15), symmetric weights
    #pragma unroll 1
    for (int r = 0; r < N_RADIAL; ++r)
    for (int s = 0; s <= r; ++s) {
        float acc = 0.0f;
        for (int c = 0; c < 6; ++c) acc += kW2[c] * M2s[r][c] * M2s[s][c];
        F[f++] = (_Float16)acc;
    }
    // c3: M3:M3 tril (15), symmetric weights
    #pragma unroll 1
    for (int r = 0; r < N_RADIAL; ++r)
    for (int s = 0; s <= r; ++s) {
        float acc = 0.0f;
        for (int c = 0; c < 10; ++c) acc += kW3[c] * M3s[r][c] * M3s[s][c];
        F[f++] = (_Float16)acc;
    }
    // c4: M2[r]_ij M2[s]_ik M2[t]_jk, tril3 (35)
    #pragma unroll 1
    for (int r = 0; r < N_RADIAL; ++r)
    #pragma unroll 1
    for (int s = 0; s <= r; ++s)
    #pragma unroll 1
    for (int t = 0; t <= s; ++t) {
        float acc = 0.0f;
        for (int i = 0; i < 3; ++i)
        for (int j = 0; j < 3; ++j)
        for (int k = 0; k < 3; ++k)
            acc += m2g(M2s, r, i, j) * m2g(M2s, s, i, k) * m2g(M2s, t, j, k);
        F[f++] = (_Float16)acc;
    }
    // c5: M1[r]_i M1[s]_j M2[t]_ij, tril2 x 5 (75)
    #pragma unroll 1
    for (int r = 0; r < N_RADIAL; ++r)
    #pragma unroll 1
    for (int s = 0; s <= r; ++s)
    #pragma unroll 1
    for (int t = 0; t < N_RADIAL; ++t) {
        float acc = 0.0f;
        for (int i = 0; i < 3; ++i)
        for (int j = 0; j < 3; ++j)
            acc += M1v[r][i] * M1v[s][j] * m2g(M2s, t, i, j);
        F[f++] = (_Float16)acc;
    }
    // c6: M3[r]_ijk M3[s]_ijl M2[t]_kl, tril2 x 5 (75)
    #pragma unroll 1
    for (int r = 0; r < N_RADIAL; ++r)
    #pragma unroll 1
    for (int s = 0; s <= r; ++s)
    #pragma unroll 1
    for (int t = 0; t < N_RADIAL; ++t) {
        float acc = 0.0f;
        #pragma unroll 1
        for (int i = 0; i < 3; ++i)
        #pragma unroll 1
        for (int j = 0; j < 3; ++j)
        for (int k = 0; k < 3; ++k)
        for (int l = 0; l < 3; ++l)
            acc += m3g(M3s, r, i, j, k) * m3g(M3s, s, i, j, l) * m2g(M2s, t, k, l);
        F[f++] = (_Float16)acc;
    }
    // c7: M3[r]_ijk M2[s]_ij M1[t]_k, full 5x5x5 (125)
    #pragma unroll 1
    for (int r = 0; r < N_RADIAL; ++r)
    #pragma unroll 1
    for (int s = 0; s < N_RADIAL; ++s)
    #pragma unroll 1
    for (int t = 0; t < N_RADIAL; ++t) {
        float acc = 0.0f;
        for (int i = 0; i < 3; ++i)
        for (int j = 0; j < 3; ++j)
        for (int k = 0; k < 3; ++k)
            acc += m3g(M3s, r, i, j, k) * m2g(M2s, s, i, j) * M1v[t][k];
        F[f++] = (_Float16)acc;
    }
    // pad K 360 -> 384 with zeros
    for (; f < K1_PAD; ++f) F[f] = (_Float16)0.0f;
}

// ---------------- weight transpose+convert: W (K x N) f32 -> Wt (N x Kpad) f16 ----------------
__global__ void mtp_convW_kernel(const float* __restrict__ W, _Float16* __restrict__ Wt,
                                 int K, int Kpad, int N)
{
    int idx = blockIdx.x * blockDim.x + threadIdx.x;
    int total = N * Kpad;
    if (idx >= total) return;
    int n = idx / Kpad;
    int k = idx - n * Kpad;
    Wt[idx] = (k < K) ? (_Float16)W[(size_t)k * N + n] : (_Float16)0.0f;
}

// ---------------- WMMA GEMM with TDM-staged B tile in LDS ----------------
// C = act( (A @ B) * scale + 0.1*bias )
// A : M x K row-major f16 (K multiple of 64), streamed from global w/ ping-pong reg buffers.
// Bt: N x K row-major f16 (B col-major). Per block, the TDM loads a 64-row x K tile of Bt
//   into LDS (tensor_load_to_lds, TENSORcnt), with D# pad_interval/pad_amount inserting 32B
//   of padding per interval so the LDS row pitch is NOT a multiple of the 256B bank row.
// Each wave computes a 32x64 strip: 2 A frags x 4 B frags -> 8 v_wmma_f32_16x16x32_f16/k-step.
// k-loop unrolled by 2 with ping-pong A buffers (no inter-buffer copies -> no mov/nop chains);
// all 4 B frag ds_loads of a step issue before the 8 WMMAs so DScnt counts down instead of
// hitting zero-waits.  EXEC stays all-ones through WMMA (loads clamped; only stores guarded).
template<int PITCH, int SEG_SHIFT, int SEG_BYTES>
__global__ __launch_bounds__(256) void mtp_gemm_tdm_wmma_kernel(
    const _Float16* __restrict__ A, const _Float16* __restrict__ Bt,
    const float* __restrict__ bias, _Float16* __restrict__ C,
    int M, int K, int N, float scale, int act,
    int pad_int_code,   // D# pad_interval code (dwords = 2^(code+1))
    int pad_amt_code)   // D# pad_amount code  (dwords = code+1)
{
    extern __shared__ char smem[];
    const int lane = threadIdx.x & 31;
    const int wave = threadIdx.x >> 5;
    const int n0   = blockIdx.y * 64;

    // ---- TDM: stage B tile [n0, n0+64) x K into LDS (one wave issues; EXEC ignored by TDM)
    if (wave == 0) {
        unsigned long long ga = (unsigned long long)(uintptr_t)(Bt + (size_t)n0 * K);
        u32x4 g0;
        g0.x = 1u;                                      // count=1, user descriptor
        g0.y = (unsigned)(uintptr_t)smem;               // lds_addr (byte offset in LDS aperture)
        g0.z = (unsigned)(ga & 0xFFFFFFFFull);          // global_addr[31:0]
        g0.w = (unsigned)((ga >> 32) & 0x01FFFFFFull)   // global_addr[56:32]
             | 0x80000000u;                             // type=2 ("image") at bits 127:126
        i32x8 g1;
        g1[0] = (1 << 16)                               // data_size: 1 -> 2 bytes
              | (1 << 20)                               // pad_enable
              | (pad_int_code << 22) | (pad_amt_code << 25);
        g1[1] = (K & 0xFFFF) << 16;                     // tensor_dim0[15:0] @bit48
        g1[2] = (K >> 16) | ((N & 0xFFFF) << 16);       // tensor_dim0[31:16] | tensor_dim1[15:0]
        g1[3] = (N >> 16) | (K << 16);                  // tensor_dim1[31:16] | tile_dim0
        g1[4] = 64;                                     // tile_dim1=64, tile_dim2=0
        g1[5] = K;                                      // tensor_dim0_stride[31:0]
        g1[6] = 0;                                      // stride hi, dim1_stride lo (unused 2D)
        g1[7] = 0;
        i32x4 gz = {0, 0, 0, 0};                        // groups 2/3 unused (2D tensor)
#if defined(__clang_major__) && (__clang_major__ >= 23)
        i32x8 gz8 = {0, 0, 0, 0, 0, 0, 0, 0};
        __builtin_amdgcn_tensor_load_to_lds(g0, g1, gz, gz, gz8, 0);
#else
        __builtin_amdgcn_tensor_load_to_lds(g0, g1, gz, gz, 0);
#endif
        __builtin_amdgcn_s_wait_tensorcnt(0);
    }
    __syncthreads();

    // ---- per-wave output strip: rows [mbase, mbase+32), cols [n0, n0+64)
    const int mbase = (blockIdx.x * 8 + wave) * 32;
    const int mA0 = min(mbase +      (lane & 15), M - 1);
    const int mA1 = min(mbase + 16 + (lane & 15), M - 1);
    // A frag (16x32 f16): elem e -> k = 16*(L>>4) + e
    const _Float16* Arow0 = A + (size_t)mA0 * K + ((lane >> 4) << 4);
    const _Float16* Arow1 = A + (size_t)mA1 * K + ((lane >> 4) << 4);

    // B frag from LDS: lane L -> col n_local, lanes 0-15 cover k0..k0+15, 16-31 k0+16..k0+31
    const int kbOff   = (lane & 16) ? 16 : 0;
    const int segMask = (1 << SEG_SHIFT) - 1;
    int rowb[4];
    #pragma unroll
    for (int c = 0; c < 4; ++c) rowb[c] = (c * 16 + (lane & 15)) * PITCH;

    v8f acc[2][4];
    #pragma unroll
    for (int ms = 0; ms < 2; ++ms)
        #pragma unroll
        for (int c = 0; c < 4; ++c)
            #pragma unroll
            for (int e = 0; e < 8; ++e) acc[ms][c][e] = 0.0f;

    // compute one 32-wide k-step: 8 ds_load_b128 up front, then 8 WMMAs
    auto step = [&](int k, const Frag16& fa0, const Frag16& fa1) {
        const int k0   = k + kbOff;
        const int boff = ((k0 >> SEG_SHIFT) * SEG_BYTES) + ((k0 & segMask) * 2);
        Frag16 fb[4];
        #pragma unroll
        for (int c = 0; c < 4; ++c) {
            const char* bp = smem + rowb[c] + boff;
            fb[c].q[0] = *(const uint4*)(bp);
            fb[c].q[1] = *(const uint4*)(bp + 16);
        }
        #pragma unroll
        for (int c = 0; c < 4; ++c) {
            acc[0][c] = __builtin_amdgcn_wmma_f32_16x16x32_f16(
                false, fa0.v, false, fb[c].v, (short)0, acc[0][c], false, false);
            acc[1][c] = __builtin_amdgcn_wmma_f32_16x16x32_f16(
                false, fa1.v, false, fb[c].v, (short)0, acc[1][c], false, false);
        }
    };
    auto loadA = [&](Frag16& f0, Frag16& f1, int kk) {
        f0.q[0] = *(const uint4*)(Arow0 + kk); f0.q[1] = *(const uint4*)(Arow0 + kk + 8);
        f1.q[0] = *(const uint4*)(Arow1 + kk); f1.q[1] = *(const uint4*)(Arow1 + kk + 8);
    };

    // k-loop unrolled x2 with ping-pong A buffers (K is a multiple of 64: 384 or 512)
    Frag16 pa0, pa1, qa0, qa1;
    loadA(pa0, pa1, 0);
    for (int k = 0; k < K; k += 64) {
        if (k + 32 < K) loadA(qa0, qa1, k + 32);   // prefetch (LOADcnt) overlaps WMMA+LDS
        step(k, pa0, pa1);
        if (k + 64 < K) loadA(pa0, pa1, k + 64);
        step(k + 32, qa0, qa1);
    }

    // C/D layout: lane L -> n = L&15; VGPR v -> m = v + 8*(L>>4)
    const int mhi = (lane >> 4) << 3;
    #pragma unroll
    for (int ms = 0; ms < 2; ++ms)
    #pragma unroll
    for (int c = 0; c < 4; ++c) {
        int col = n0 + c * 16 + (lane & 15);
        float bb = 0.1f * bias[col];
        #pragma unroll
        for (int v = 0; v < 8; ++v) {
            int m = mbase + ms * 16 + mhi + v;
            if (m < M) {
                float x = acc[ms][c][v] * scale + bb;
                if (act) x = x / (1.0f + __expf(-x));   // swish
                C[(size_t)m * N + col] = (_Float16)x;
            }
        }
    }
}

// ---------------- final layer: out[a] = mask * (h2[a,:].W3 / sqrt(512) + 0.1*b3) ----------------
__global__ __launch_bounds__(256) void mtp_out_kernel(
    const _Float16* __restrict__ H2, const float* __restrict__ W3,
    const float* __restrict__ b3, const int* __restrict__ Z,
    float* __restrict__ out)
{
    int gid  = blockIdx.x * blockDim.x + threadIdx.x;
    int atom = gid >> 5;
    int lane = gid & 31;
    if (atom >= N_ATOMS) return;
    const _Float16* h = H2 + (size_t)atom * UNITS;
    float s = 0.0f;
    #pragma unroll
    for (int t = 0; t < UNITS / 32; ++t) {
        int c = lane + 32 * t;
        s += (float)h[c] * W3[c];
    }
    #pragma unroll
    for (int off = 16; off > 0; off >>= 1) s += __shfl_xor(s, off, 32);
    if (lane == 0)
        out[atom] = (Z[atom] > 0) ? (s * RSQ512 + 0.1f * b3[0]) : 0.0f;
}

// ---------------- launcher ----------------
extern "C" void kernel_launch(void* const* d_in, const int* in_sizes, int n_in,
                              void* d_out, int out_size, void* d_ws, size_t ws_size,
                              hipStream_t stream)
{
    const float* R    = (const float*)d_in[0];
    const int*   Z    = (const int*  )d_in[1];
    const int*   nbr  = (const int*  )d_in[2];
    /* d_in[3] = box (unused, zeros) */
    const float* offs = (const float*)d_in[4];
    const float* emb  = (const float*)d_in[5];
    const float* W1   = (const float*)d_in[6];
    const float* b1   = (const float*)d_in[7];
    const float* W2   = (const float*)d_in[8];
    const float* b2   = (const float*)d_in[9];
    const float* W3   = (const float*)d_in[10];
    const float* b3   = (const float*)d_in[11];
    float* out = (float*)d_out;

    // workspace carve-up (all chunks 256B aligned)
    char* ws = (char*)d_ws;
    size_t off = 0;
    float*     moments = (float*)    (ws + off); off += (size_t)N_ATOMS * MOM_PER_ATOM * 4;   // 8.0 MB
    _Float16*  feats   = (_Float16*) (ws + off); off += (size_t)N_ATOMS * K1_PAD * 2;         // 15.36 MB
    _Float16*  W1t     = (_Float16*) (ws + off); off += (size_t)UNITS * K1_PAD * 2;           // 0.39 MB
    _Float16*  h1      = (_Float16*) (ws + off); off += (size_t)N_ATOMS * UNITS * 2;          // 20.48 MB
    _Float16*  W2t     = (_Float16*) (ws + off); off += (size_t)UNITS * UNITS * 2;            // 0.52 MB
    _Float16*  h2      = (_Float16*) (ws + off); off += (size_t)N_ATOMS * UNITS * 2;          // 20.48 MB

    // allow >64KB dynamic LDS for the GEMM2 tile (67584B); harmless if already allowed
    static bool attr_set = false;  // host-side only; does not affect device work (idempotent)
    if (!attr_set) {
        hipFuncSetAttribute((const void*)&mtp_gemm_tdm_wmma_kernel<864, 7, 288>,
                            hipFuncAttributeMaxDynamicSharedMemorySize, 68 * 1024);
        hipFuncSetAttribute((const void*)&mtp_gemm_tdm_wmma_kernel<1056, 9, 1056>,
                            hipFuncAttributeMaxDynamicSharedMemorySize, 68 * 1024);
        attr_set = true;
    }

    // 0) zero moment accumulator
    int nmom = N_ATOMS * MOM_PER_ATOM;
    mtp_zero_kernel<<<(nmom + 255) / 256, 256, 0, stream>>>(moments, nmom);

    // 1) pair stage
    mtp_pair_kernel<<<(N_PAIRS + 255) / 256, 256, 0, stream>>>(R, Z, nbr, offs, emb, moments);

    // 2) contractions -> f16 features (padded K=384)
    mtp_contract_kernel<<<(N_ATOMS + 127) / 128, 128, 0, stream>>>(moments, feats);

    // 3) weight convert/transpose (independent of 0-2)
    {
        int t1 = UNITS * K1_PAD;
        mtp_convW_kernel<<<(t1 + 255) / 256, 256, 0, stream>>>(W1, W1t, N_FEAT, K1_PAD, UNITS);
        int t2 = UNITS * UNITS;
        mtp_convW_kernel<<<(t2 + 255) / 256, 256, 0, stream>>>(W2, W2t, UNITS, UNITS, UNITS);
    }

    // 4) MLP via TDM + WMMA.  Block: 8 waves x (32x64) = 256x64 tile.
    dim3 ggrid((N_ATOMS + 255) / 256, UNITS / 64);   // (79, 8)
    // GEMM1: K=384. rows 768B = 3 x 256B intervals; +32B pad each -> pitch 864B (not mult of 256B)
    //   pad_interval code 5 (64 dwords = 256B), pad_amount code 7 (8 dwords = 32B)
    mtp_gemm_tdm_wmma_kernel<864, 7, 288><<<ggrid, 256, 64 * 864, stream>>>(
        feats, W1t, b1, h1, N_ATOMS, K1_PAD, UNITS, RSQ360, 1, /*pad_int*/5, /*pad_amt*/7);
    // GEMM2: K=512. rows 1024B = 1 x 1024B interval; +32B pad -> pitch 1056B
    //   pad_interval code 7 (256 dwords = 1024B), pad_amount code 7 (8 dwords = 32B)
    mtp_gemm_tdm_wmma_kernel<1056, 9, 1056><<<ggrid, 256, 64 * 1056, stream>>>(
        h1, W2t, b2, h2, N_ATOMS, UNITS, UNITS, RSQ512, 1, /*pad_int*/7, /*pad_amt*/7);

    // 5) final dot + mask
    mtp_out_kernel<<<(N_ATOMS * 32) / 256, 256, 0, stream>>>(h2, W3, b3, Z, out);
}
